// Nystromformer_52982716564168
// MI455X (gfx1250) — compile-verified, matrix operand order checked by reading, not run on hardware
//
#include <hip/hip_runtime.h>
#include <math.h>

// ---------------------------------------------------------------------------
// Nystromformer forward for MI455X (gfx1250, wave32, WMMA 16x16x32 f16).
// Every GEMM operand pair is (A row-major MxK, B row-major NxK) so both sides
// use 128-bit contiguous loads; each wave computes a 32x64 tile (8 WMMAs per
// 32-deep K chunk).
// ---------------------------------------------------------------------------
#define B_    16
#define L_    1000
#define D_    512
#define H_    8
#define DEPTH_ 6
#define M_    256
#define FF_   2048
#define DH_   64
#define NPAD  1024
#define PAD_  24
#define KC_   33
#define SCALE_ 0.125f   // DH^-0.5

typedef _Float16 h16;
typedef __attribute__((ext_vector_type(8)))  _Float16 v8h;
typedef __attribute__((ext_vector_type(16))) _Float16 v16h;
typedef __attribute__((ext_vector_type(8)))  float    v8f;

// Load one 16x32 f16 fragment (A or B^T side) per CDNA5 ISA wave32
// layout: lanes 0-15 hold K 0-7 & 16-23 of row `lane`, lanes 16-31 hold
// K 8-15 & 24-31 (caller pre-offsets the pointer by hlf*8).
__device__ __forceinline__ v16h ldfrag(const h16* p) {
  v8h lo = *(const v8h*)(p);
  v8h hi = *(const v8h*)(p + 16);
  return __builtin_shufflevector(lo, hi, 0,1,2,3,4,5,6,7,8,9,10,11,12,13,14,15);
}

// ---------------------------------------------------------------------------
// Batched WMMA GEMM: C[Mr,Nc] = alpha * A[Mr,K] * B^T, B stored row-major NxK.
// One wave -> 32x64 output tile (2x4 subtiles), 4 waves per block.
// ---------------------------------------------------------------------------
__global__ __launch_bounds__(128)
void k_gemm(const h16* __restrict__ A, const h16* __restrict__ Bm,
            float* __restrict__ C, int Mr, int Nc, int K,
            size_t sA, size_t sB, size_t sC, float alpha) {
  const int tilesN = Nc >> 6;               // 64-wide tiles
  const int tiles  = (Mr >> 5) * tilesN;    // 32-tall tiles
  const int tile   = blockIdx.x * 4 + threadIdx.y;  // uniform per wave
  if (tile >= tiles) return;
  const int tm = tile / tilesN;
  const int tn = tile % tilesN;
  const int lane = threadIdx.x;
  const int hlf  = lane >> 4;
  const int r    = lane & 15;

  const h16* Ab = A  + (size_t)blockIdx.z * sA;
  const h16* Bb = Bm + (size_t)blockIdx.z * sB;
  float*     Cb = C  + (size_t)blockIdx.z * sC;

  const h16* ar0 = Ab + (size_t)(tm * 32 + r) * K + hlf * 8;
  const h16* ar1 = ar0 + (size_t)16 * K;
  const h16* br0 = Bb + (size_t)(tn * 64 + r) * K + hlf * 8;

  v8f acc[8] = {};   // acc[mi*4 + ni]

  for (int kb = 0; kb < K; kb += 32) {
    __builtin_prefetch(ar0 + kb + 256, 0, 0);   // gfx1250 global_prefetch_b8
    __builtin_prefetch(br0 + kb + 256, 0, 0);
    v16h av[2], bv[4];
    av[0] = ldfrag(ar0 + kb);
    av[1] = ldfrag(ar1 + kb);
#pragma unroll
    for (int ni = 0; ni < 4; ++ni)
      bv[ni] = ldfrag(br0 + (size_t)(ni * 16) * K + kb);
#pragma unroll
    for (int mi = 0; mi < 2; ++mi)
#pragma unroll
      for (int ni = 0; ni < 4; ++ni)
        acc[mi * 4 + ni] = __builtin_amdgcn_wmma_f32_16x16x32_f16(
            false, av[mi], false, bv[ni], (short)0, acc[mi * 4 + ni],
            false, false);
  }

#pragma unroll
  for (int mi = 0; mi < 2; ++mi)
#pragma unroll
    for (int ni = 0; ni < 4; ++ni) {
      const int cn = tn * 64 + ni * 16 + r;
#pragma unroll
      for (int rr = 0; rr < 8; ++rr) {
        const int cm = tm * 32 + mi * 16 + rr + hlf * 8;
        Cb[(size_t)cm * Nc + cn] = alpha * acc[mi * 4 + ni][rr];
      }
    }
}

// ---------------------------------------------------------------------------
// Elementwise / reduction kernels
// ---------------------------------------------------------------------------
__global__ void k_embed(const int* __restrict__ x, const float* __restrict__ emb,
                        const float* __restrict__ pos, float* __restrict__ h,
                        long total) {
  for (long i = (long)blockIdx.x * blockDim.x + threadIdx.x; i < total;
       i += (long)gridDim.x * blockDim.x) {
    long d  = i % D_;
    long bl = i / D_;
    long l  = bl % L_;
    int tok = x[bl];
    h[i] = emb[(size_t)tok * D_ + d] + pos[(size_t)l * D_ + d];
  }
}

// LayerNorm over D=512; writes f16 rows, zero-fills `pad` leading rows per batch
__global__ void k_layernorm(const float* __restrict__ in, h16* __restrict__ out,
                            const float* __restrict__ sc, const float* __restrict__ bi,
                            int rowsIn, int rowsOut, int pad) {
  const int row = blockIdx.x;
  const int b = row / rowsOut, r = row % rowsOut;
  h16* orow = out + ((size_t)b * rowsOut + r) * D_;
  const int t = threadIdx.x;  // 256
  if (r < pad) {
    for (int d = t; d < D_; d += 256) orow[d] = (h16)0.f;
    return;
  }
  const float* irow = in + ((size_t)b * rowsIn + (r - pad)) * D_;
  __shared__ float red[256];
  float s = 0.f;
  for (int d = t; d < D_; d += 256) s += irow[d];
  red[t] = s; __syncthreads();
  for (int k = 128; k > 0; k >>= 1) { if (t < k) red[t] += red[t + k]; __syncthreads(); }
  const float mu = red[0] / (float)D_;
  __syncthreads();
  float vs = 0.f;
  for (int d = t; d < D_; d += 256) { float u = irow[d] - mu; vs += u * u; }
  red[t] = vs; __syncthreads();
  for (int k = 128; k > 0; k >>= 1) { if (t < k) red[t] += red[t + k]; __syncthreads(); }
  const float rinv = rsqrtf(red[0] / (float)D_ + 1e-5f);
  for (int d = t; d < D_; d += 256)
    orow[d] = (h16)((irow[d] - mu) * rinv * sc[d] + bi[d]);
}

// qkv (B,N,3D) f32 -> q*SCALE,k in (B,H,N,DH) f16; v transposed (B,H,DH,N) f16
// (so v can be the NT-B operand of attn3@v) + v f32 (B,H,N,DH) for the conv.
__global__ void k_split_qkv(const float* __restrict__ qkv, h16* __restrict__ q,
                            h16* __restrict__ k, h16* __restrict__ vt,
                            float* __restrict__ v32) {
  const long total = (long)B_ * NPAD * D_;
  for (long i = (long)blockIdx.x * blockDim.x + threadIdx.x; i < total;
       i += (long)gridDim.x * blockDim.x) {
    long d = i % D_;
    long n = (i / D_) % NPAD;
    long b = i / ((long)D_ * NPAD);
    int hh = (int)(d / DH_), dh = (int)(d % DH_);
    size_t src = ((size_t)b * NPAD + n) * (3 * D_);
    size_t bh  = (size_t)b * H_ + hh;
    size_t dst = (bh * NPAD + n) * DH_ + dh;
    float qv = qkv[src + d] * SCALE_;
    float kv = qkv[src + D_ + d];
    float vv = qkv[src + 2 * D_ + d];
    q[dst] = (h16)qv; k[dst] = (h16)kv;
    vt[(bh * DH_ + dh) * NPAD + n] = (h16)vv;
    v32[dst] = vv;
  }
}

// Landmarks: mean over 4 consecutive sequence positions (N = M*4)
__global__ void k_landmarks(const float* __restrict__ qkv, h16* __restrict__ ql,
                            h16* __restrict__ kl) {
  const int idx = blockIdx.x;              // B*H*M
  const int dh = threadIdx.x;              // 64
  const int m = idx % M_;
  const int hh = (idx / M_) % H_;
  const int b = idx / (M_ * H_);
  float qs = 0.f, ks = 0.f;
#pragma unroll
  for (int j = 0; j < 4; ++j) {
    size_t row = ((size_t)b * NPAD + (m * 4 + j)) * (3 * D_);
    qs += qkv[row + hh * DH_ + dh];
    ks += qkv[row + D_ + hh * DH_ + dh];
  }
  size_t dst = (((size_t)b * H_ + hh) * M_ + m) * DH_ + dh;
  ql[dst] = (h16)(qs * 0.25f * SCALE_);
  kl[dst] = (h16)(ks * 0.25f);
}

// Row softmax, f32 in -> f16 out. One block (256 threads) per row, cols<=1024.
__global__ void k_softmax(const float* __restrict__ in, h16* __restrict__ out, int cols) {
  const size_t row = blockIdx.x;
  const float* x = in + row * (size_t)cols;
  h16* o = out + row * (size_t)cols;
  __shared__ float red[256];
  const int t = threadIdx.x;
  float mx = -3.4e38f;
  for (int c = t; c < cols; c += 256) mx = fmaxf(mx, x[c]);
  red[t] = mx; __syncthreads();
  for (int k = 128; k > 0; k >>= 1) { if (t < k) red[t] = fmaxf(red[t], red[t + k]); __syncthreads(); }
  mx = red[0]; __syncthreads();
  float ev[4];
  float s = 0.f;
  int cnt = 0;
  for (int c = t; c < cols; c += 256) { float e = __expf(x[c] - mx); ev[cnt++] = e; s += e; }
  red[t] = s; __syncthreads();
  for (int k = 128; k > 0; k >>= 1) { if (t < k) red[t] += red[t + k]; __syncthreads(); }
  const float rs = 1.f / red[0];
  cnt = 0;
  for (int c = t; c < cols; c += 256) o[c] = (h16)(ev[cnt++] * rs);
}

// z0 = x^T / (max rowsum(|x|) * max colsum(|x|)); writes z (row-major, A role)
// and z^T (row-major, NT-B role). Note zT = x * denom (contiguous copy).
__global__ void k_pinv_init(const h16* __restrict__ x, h16* __restrict__ z,
                            h16* __restrict__ zt) {
  const size_t base = (size_t)blockIdx.x * M_ * M_;
  const int t = threadIdx.x;  // 256
  float rsum = 0.f, csum = 0.f;
  for (int j = 0; j < M_; ++j) {
    rsum += fabsf((float)x[base + (size_t)t * M_ + j]);
    csum += fabsf((float)x[base + (size_t)j * M_ + t]);
  }
  __shared__ float rmax[256], cmax[256];
  rmax[t] = rsum; cmax[t] = csum; __syncthreads();
  for (int k = 128; k > 0; k >>= 1) {
    if (t < k) { rmax[t] = fmaxf(rmax[t], rmax[t + k]); cmax[t] = fmaxf(cmax[t], cmax[t + k]); }
    __syncthreads();
  }
  const float denom = 1.f / (rmax[0] * cmax[0]);
  for (int j = 0; j < M_; ++j) {
    z [base + (size_t)t * M_ + j] = (h16)((float)x[base + (size_t)j * M_ + t] * denom);
    zt[base + (size_t)t * M_ + j] = (h16)((float)x[base + (size_t)t * M_ + j] * denom);
  }
}

// outT = (c*I - in)^T = c*I - in^T  (batched 256x256), f32 -> f16 transposed
__global__ void k_diag_tr(const float* __restrict__ in, h16* __restrict__ outT,
                          float c, long total) {
  for (long i = (long)blockIdx.x * blockDim.x + threadIdx.x; i < total;
       i += (long)gridDim.x * blockDim.x) {
    long mat = i / ((long)M_ * M_);
    long rem = i % ((long)M_ * M_);
    int rr = (int)(rem / M_), cc = (int)(rem % M_);
    outT[(size_t)mat * M_ * M_ + (size_t)cc * M_ + rr] =
        (h16)(((rr == cc) ? c : 0.f) - in[i]);
  }
}

__global__ void k_f32_to_f16(const float* __restrict__ in, h16* __restrict__ out, long total) {
  for (long i = (long)blockIdx.x * blockDim.x + threadIdx.x; i < total;
       i += (long)gridDim.x * blockDim.x)
    out[i] = (h16)in[i];
}

// batched RxC f32 -> CxR f16 transpose-cast (out[c*R + r] = in[r*C + c])
__global__ void k_cvt_tr(const float* __restrict__ in, h16* __restrict__ out,
                         int R, int C, long total) {
  for (long i = (long)blockIdx.x * blockDim.x + threadIdx.x; i < total;
       i += (long)gridDim.x * blockDim.x) {
    long rc = (long)R * C;
    long b = i / rc, rem = i % rc;
    int rr = (int)(rem / C), cc = (int)(rem % C);
    out[(size_t)b * rc + (size_t)cc * R + rr] = (h16)in[i];
  }
}

// batched 256x256: z (normal) and z^T, both f16, from f32
__global__ void k_cvt_dual(const float* __restrict__ in, h16* __restrict__ outN,
                           h16* __restrict__ outT, long total) {
  for (long i = (long)blockIdx.x * blockDim.x + threadIdx.x; i < total;
       i += (long)gridDim.x * blockDim.x) {
    long mat = i / ((long)M_ * M_);
    long rem = i % ((long)M_ * M_);
    int rr = (int)(rem / M_), cc = (int)(rem % M_);
    float v = in[i];
    outN[i] = (h16)v;
    outT[(size_t)mat * M_ * M_ + (size_t)cc * M_ + rr] = (h16)v;
  }
}

// o += depthwise conv over sequence (kernel 33, per head), zero-padded
__global__ void k_conv_add(const float* __restrict__ v, const float* __restrict__ ck,
                           float* __restrict__ o, int layer) {
  const int n = blockIdx.x;       // NPAD
  const int bh = blockIdx.y;      // B*H
  const int dh = threadIdx.x;     // 64
  const int hh = bh % H_;
  const float* kk = ck + ((size_t)layer * H_ + hh) * KC_;
  const float* vb = v + (size_t)bh * NPAD * DH_;
  float acc = 0.f;
#pragma unroll
  for (int t = 0; t < KC_; ++t) {
    int nn = n + t - KC_ / 2;
    if (nn >= 0 && nn < NPAD) acc += vb[(size_t)nn * DH_ + dh] * kk[t];
  }
  o[(size_t)bh * NPAD * DH_ + (size_t)n * DH_ + dh] += acc;
}

// (B,H,N,DH) f32 -> (B,N,D) f16
__global__ void k_merge(const float* __restrict__ o, h16* __restrict__ merged, long total) {
  for (long i = (long)blockIdx.x * blockDim.x + threadIdx.x; i < total;
       i += (long)gridDim.x * blockDim.x) {
    long d = i % D_;
    long n = (i / D_) % NPAD;
    long b = i / ((long)D_ * NPAD);
    int hh = (int)(d / DH_), dh = (int)(d % DH_);
    merged[i] = (h16)o[(((size_t)b * H_ + hh) * NPAD + n) * DH_ + dh];
  }
}

// h[b,l,:] += p[b, l+PAD, :] + bias  (drop left padding)
__global__ void k_resid_slice(float* __restrict__ h, const float* __restrict__ p,
                              const float* __restrict__ bias, long total) {
  for (long i = (long)blockIdx.x * blockDim.x + threadIdx.x; i < total;
       i += (long)gridDim.x * blockDim.x) {
    long d = i % D_;
    long l = (i / D_) % L_;
    long b = i / ((long)D_ * L_);
    h[i] += p[((size_t)b * NPAD + l + PAD_) * D_ + d] + bias[d];
  }
}

__global__ void k_gelu(const float* __restrict__ in, const float* __restrict__ b1,
                       h16* __restrict__ out, long total) {
  for (long i = (long)blockIdx.x * blockDim.x + threadIdx.x; i < total;
       i += (long)gridDim.x * blockDim.x) {
    int c = (int)(i % FF_);
    float xv = in[i] + b1[c];
    out[i] = (h16)(0.5f * xv * (1.f + erff(xv * 0.70710678118f)));
  }
}

__global__ void k_resid_ff(float* __restrict__ h, const float* __restrict__ f,
                           const float* __restrict__ b2, long total) {
  for (long i = (long)blockIdx.x * blockDim.x + threadIdx.x; i < total;
       i += (long)gridDim.x * blockDim.x) {
    long d = i % D_;
    h[i] += f[i] + b2[d];
  }
}

__global__ void k_final(const float* __restrict__ h, const float* __restrict__ fw,
                        const float* __restrict__ fb, float* __restrict__ out) {
  const int b = blockIdx.x;
  const int t = threadIdx.x;  // 256
  const float* hb = h + (size_t)b * L_ * D_;
  float s = 0.f;
  for (long i = t; i < (long)L_ * D_; i += 256) s += hb[i] * fw[i];
  __shared__ float red[256];
  red[t] = s; __syncthreads();
  for (int k = 128; k > 0; k >>= 1) { if (t < k) red[t] += red[t + k]; __syncthreads(); }
  if (t == 0) out[b] = red[0] + fb[0];
}

// ---------------------------------------------------------------------------
// Host orchestration
// ---------------------------------------------------------------------------
static inline unsigned gs_blocks(long total) {
  long b = (total + 255) / 256;
  return (unsigned)b;
}

extern "C" void kernel_launch(void* const* d_in, const int* in_sizes, int n_in,
                              void* d_out, int out_size, void* d_ws, size_t ws_size,
                              hipStream_t stream) {
  (void)in_sizes; (void)n_in; (void)out_size; (void)ws_size;
  const int*   x_tok  = (const int*)  d_in[0];
  const float* enc_emb= (const float*)d_in[1];
  const float* pos_emb= (const float*)d_in[2];
  const float* ln1_s  = (const float*)d_in[3];
  const float* ln1_b  = (const float*)d_in[4];
  const float* qkv_w  = (const float*)d_in[5];
  const float* aow    = (const float*)d_in[6];
  const float* aob    = (const float*)d_in[7];
  const float* convk  = (const float*)d_in[8];
  const float* ln2_s  = (const float*)d_in[9];
  const float* ln2_b  = (const float*)d_in[10];
  const float* ffw1   = (const float*)d_in[11];
  const float* ffb1   = (const float*)d_in[12];
  const float* ffw2   = (const float*)d_in[13];
  const float* ffb2   = (const float*)d_in[14];
  const float* fin_w  = (const float*)d_in[15];
  const float* fin_b  = (const float*)d_in[16];
  float* out = (float*)d_out;

  // ---- workspace carve (256B aligned bump allocator) ----
  char* ws = (char*)d_ws;
  size_t off = 0;
  auto alloc = [&](size_t bytes) -> char* {
    char* p = ws + off;
    off = (off + bytes + 255) & ~(size_t)255;
    return p;
  };
  float* h32   = (float*)alloc((size_t)B_ * L_ * D_ * 4);
  h16* wqt16   = (h16*)alloc((size_t)DEPTH_ * D_ * 3 * D_ * 2);   // (3D x D) per layer
  h16* wot16   = (h16*)alloc((size_t)DEPTH_ * D_ * D_ * 2);       // (D x D)
  h16* w1t16   = (h16*)alloc((size_t)DEPTH_ * D_ * FF_ * 2);      // (FF x D)
  h16* w2t16   = (h16*)alloc((size_t)DEPTH_ * FF_ * D_ * 2);      // (D x FF)
  h16* xln16   = (h16*)alloc((size_t)B_ * NPAD * D_ * 2);         // also merged16 / LN2 out
  float* qkv32 = (float*)alloc((size_t)B_ * NPAD * 3 * D_ * 4);
  h16* qf16    = (h16*)alloc((size_t)B_ * H_ * NPAD * DH_ * 2);
  h16* kf16    = (h16*)alloc((size_t)B_ * H_ * NPAD * DH_ * 2);
  h16* vt16    = (h16*)alloc((size_t)B_ * H_ * DH_ * NPAD * 2);   // transposed v
  float* vf32  = (float*)alloc((size_t)B_ * H_ * NPAD * DH_ * 4);
  h16* ql16    = (h16*)alloc((size_t)B_ * H_ * M_ * DH_ * 2);
  h16* kl16    = (h16*)alloc((size_t)B_ * H_ * M_ * DH_ * 2);
  float* sc32  = (float*)alloc((size_t)B_ * H_ * NPAD * M_ * 4);  // scores1/3, w1, ff1
  h16* s1_16   = (h16*)alloc((size_t)B_ * H_ * NPAD * M_ * 2);
  h16* s3_16   = (h16*)alloc((size_t)B_ * H_ * NPAD * M_ * 2);    // also w1_16 / gelu16
  float* a2_32 = (float*)alloc((size_t)B_ * H_ * M_ * M_ * 4);    // also ff2 out
  h16* x2_16   = (h16*)alloc((size_t)B_ * H_ * M_ * M_ * 2);
  h16* z16     = (h16*)alloc((size_t)B_ * H_ * M_ * M_ * 2);      // z (A role)
  h16* z16t    = (h16*)alloc((size_t)B_ * H_ * M_ * M_ * 2);      // z^T (B role)
  float* z32   = (float*)alloc((size_t)B_ * H_ * M_ * M_ * 4);
  h16* xz16    = (h16*)alloc((size_t)B_ * H_ * M_ * M_ * 2);
  h16* t16t    = (h16*)alloc((size_t)B_ * H_ * M_ * M_ * 2);      // (cI - xz)^T
  float* u32   = (float*)alloc((size_t)B_ * H_ * M_ * M_ * 4);
  float* kv32  = (float*)alloc((size_t)B_ * H_ * M_ * DH_ * 4);
  h16* kv16t   = (h16*)alloc((size_t)B_ * H_ * DH_ * M_ * 2);     // kv^T (B role)
  float* o32   = (float*)alloc((size_t)B_ * H_ * NPAD * DH_ * 4);
  float* p32   = (float*)alloc((size_t)B_ * NPAD * D_ * 4);

  const int BH = B_ * H_;
  const dim3 gblk(32, 4, 1);
  auto gemm = [&](const h16* A, const h16* Bm, float* C, int Mr, int Nc, int K,
                  size_t sA, size_t sB, size_t sC, int batch, float alpha) {
    int tiles = (Mr / 32) * (Nc / 64);
    dim3 grid((unsigned)((tiles + 3) / 4), 1, (unsigned)batch);
    hipLaunchKernelGGL(k_gemm, grid, gblk, 0, stream, A, Bm, C, Mr, Nc, K, sA, sB, sC, alpha);
  };

  // ---- one-time per call: transposed f16 weight casts (N x K layouts) ----
  k_cvt_tr<<<gs_blocks((long)DEPTH_ * D_ * 3 * D_), 256, 0, stream>>>(qkv_w, wqt16, D_, 3 * D_, (long)DEPTH_ * D_ * 3 * D_);
  k_cvt_tr<<<gs_blocks((long)DEPTH_ * D_ * D_), 256, 0, stream>>>(aow, wot16, D_, D_, (long)DEPTH_ * D_ * D_);
  k_cvt_tr<<<gs_blocks((long)DEPTH_ * D_ * FF_), 256, 0, stream>>>(ffw1, w1t16, D_, FF_, (long)DEPTH_ * D_ * FF_);
  k_cvt_tr<<<gs_blocks((long)DEPTH_ * FF_ * D_), 256, 0, stream>>>(ffw2, w2t16, FF_, D_, (long)DEPTH_ * FF_ * D_);

  // ---- embedding ----
  k_embed<<<gs_blocks((long)B_ * L_ * D_), 256, 0, stream>>>(x_tok, enc_emb, pos_emb, h32, (long)B_ * L_ * D_);

  for (int i = 0; i < DEPTH_; ++i) {
    const h16* wq = wqt16 + (size_t)i * D_ * 3 * D_;
    const h16* wo = wot16 + (size_t)i * D_ * D_;
    const h16* w1 = w1t16 + (size_t)i * D_ * FF_;
    const h16* w2 = w2t16 + (size_t)i * FF_ * D_;

    // LN1 (with left pad to NPAD) -> f16
    k_layernorm<<<B_ * NPAD, 256, 0, stream>>>(h32, xln16, ln1_s + (size_t)i * D_, ln1_b + (size_t)i * D_, L_, NPAD, PAD_);
    // QKV projection: (16384 x 512) @ (512 x 1536)
    gemm(xln16, wq, qkv32, B_ * NPAD, 3 * D_, D_, 0, 0, 0, 1, 1.f);
    // split heads (+ scale q, transpose v), landmarks
    k_split_qkv<<<gs_blocks((long)B_ * NPAD * D_), 256, 0, stream>>>(qkv32, qf16, kf16, vt16, vf32);
    k_landmarks<<<B_ * H_ * M_, 64, 0, stream>>>(qkv32, ql16, kl16);

    // attn1 = softmax(q @ kl^T)   (batched 1024x256x64)
    gemm(qf16, kl16, sc32, NPAD, M_, DH_, (size_t)NPAD * DH_, (size_t)M_ * DH_, (size_t)NPAD * M_, BH, 1.f);
    k_softmax<<<BH * NPAD, 256, 0, stream>>>(sc32, s1_16, M_);
    // attn2 = softmax(ql @ kl^T)  (batched 256x256x64)
    gemm(ql16, kl16, a2_32, M_, M_, DH_, (size_t)M_ * DH_, (size_t)M_ * DH_, (size_t)M_ * M_, BH, 1.f);
    k_softmax<<<BH * M_, 256, 0, stream>>>(a2_32, x2_16, M_);
    // attn3 = softmax(ql @ k^T)   (batched 256x1024x64)
    gemm(ql16, kf16, sc32, M_, NPAD, DH_, (size_t)M_ * DH_, (size_t)NPAD * DH_, (size_t)M_ * NPAD, BH, 1.f);
    k_softmax<<<BH * M_, 256, 0, stream>>>(sc32, s3_16, NPAD);
    // kv = attn3 @ v  (batched 256x64x1024), B = v^T (64x1024)
    gemm(s3_16, vt16, kv32, M_, DH_, NPAD, (size_t)M_ * NPAD, (size_t)DH_ * NPAD, (size_t)M_ * DH_, BH, 1.f);
    k_cvt_tr<<<gs_blocks((long)BH * M_ * DH_), 256, 0, stream>>>(kv32, kv16t, M_, DH_, (long)BH * M_ * DH_);

    // Moore-Penrose pinv of attn2 (6 Newton iterations, batched 256^3 WMMA GEMMs)
    k_pinv_init<<<BH, 256, 0, stream>>>(x2_16, z16, z16t);
    const long mm = (long)BH * M_ * M_;
    const size_t smm = (size_t)M_ * M_;
    for (int it = 0; it < 6; ++it) {
      gemm(x2_16, z16t, u32, M_, M_, M_, smm, smm, smm, BH, 1.f);               // xz
      k_f32_to_f16<<<gs_blocks(mm), 256, 0, stream>>>(u32, xz16, mm);
      k_diag_tr<<<gs_blocks(mm), 256, 0, stream>>>(u32, t16t, 7.f, mm);         // A1^T
      gemm(xz16, t16t, u32, M_, M_, M_, smm, smm, smm, BH, 1.f);                // xz@A1
      k_diag_tr<<<gs_blocks(mm), 256, 0, stream>>>(u32, t16t, 15.f, mm);        // A2^T
      gemm(xz16, t16t, u32, M_, M_, M_, smm, smm, smm, BH, 1.f);                // xz@A2
      k_diag_tr<<<gs_blocks(mm), 256, 0, stream>>>(u32, t16t, 13.f, mm);        // A3^T
      gemm(z16, t16t, z32, M_, M_, M_, smm, smm, smm, BH, 0.25f);               // z = 0.25 z@A3
      k_cvt_dual<<<gs_blocks(mm), 256, 0, stream>>>(z32, z16, z16t, mm);
    }

    // w1 = attn1 @ pinv(attn2)   (batched 1024x256x256), B = z^T
    gemm(s1_16, z16t, sc32, NPAD, M_, M_, (size_t)NPAD * M_, smm, (size_t)NPAD * M_, BH, 1.f);
    k_f32_to_f16<<<gs_blocks((long)BH * NPAD * M_), 256, 0, stream>>>(sc32, s3_16, (long)BH * NPAD * M_);
    // out = w1 @ kv              (batched 1024x64x256), B = kv^T (64x256)
    gemm(s3_16, kv16t, o32, NPAD, DH_, M_, (size_t)NPAD * M_, (size_t)DH_ * M_, (size_t)NPAD * DH_, BH, 1.f);
    // += depthwise conv residual over sequence
    k_conv_add<<<dim3(NPAD, BH), 64, 0, stream>>>(vf32, convk, o32, i);
    // merge heads -> (B,N,D) f16, output projection, residual (drop pad, +bias)
    k_merge<<<gs_blocks((long)B_ * NPAD * D_), 256, 0, stream>>>(o32, xln16, (long)B_ * NPAD * D_);
    gemm(xln16, wo, p32, B_ * NPAD, D_, D_, 0, 0, 0, 1, 1.f);
    k_resid_slice<<<gs_blocks((long)B_ * L_ * D_), 256, 0, stream>>>(h32, p32, aob + (size_t)i * D_, (long)B_ * L_ * D_);

    // FFN: LN2 -> GEMM -> GELU -> GEMM -> residual
    k_layernorm<<<B_ * L_, 256, 0, stream>>>(h32, xln16, ln2_s + (size_t)i * D_, ln2_b + (size_t)i * D_, L_, L_, 0);
    gemm(xln16, w1, sc32, B_ * L_, FF_, D_, 0, 0, 0, 1, 1.f);
    k_gelu<<<gs_blocks((long)B_ * L_ * FF_), 256, 0, stream>>>(sc32, ffb1 + (size_t)i * FF_, s3_16, (long)B_ * L_ * FF_);
    gemm(s3_16, w2, a2_32, B_ * L_, D_, FF_, 0, 0, 0, 1, 1.f);
    k_resid_ff<<<gs_blocks((long)B_ * L_ * D_), 256, 0, stream>>>(h32, a2_32, ffb2 + (size_t)i * D_, (long)B_ * L_ * D_);
  }

  // final classifier: out[b] = h[b,:] . final_w + final_b
  k_final<<<B_, 256, 0, stream>>>(h32, fin_w, fin_b, out);
}
